// MultiHeadSelfAttention_57019985822433
// MI455X (gfx1250) — matrix operand order repaired and checked by reading, not compiled, not run on hardware
//
#include <hip/hip_runtime.h>
#include <hip/hip_bf16.h>

// ---------------------------------------------------------------------------
// RoPE multi-head self-attention for MI455X (gfx1250, wave32, WMMA).
// B=4, DIM=512, H=W=64 -> N=4096, HEADS=8, head_dim=64.
// All GEMMs via v_wmma_f32_16x16x32_bf16 (bf16 A/B, fp32 accum).
// GEMM kernels: 64x256 blocks, 8 waves as 2(M)x4(N), double-buffered LDS with
// register-pipelined global prefetch (1 barrier per k-step, loads hidden
// under the WMMA chain). Packed b128 LDS staging stores.
// Layouts keep every WMMA fragment lane-contiguous (2x b128 per lane):
//   Q,K ws: [b][h][n][d], V ws: [b][h][d][n], O ws: [b][n][c].
// Workspace: 4 arrays x 16.78 MB bf16 = 67.1 MB.
// ---------------------------------------------------------------------------

typedef __attribute__((ext_vector_type(16))) __bf16 v16bf;
typedef __attribute__((ext_vector_type(8)))  __bf16 v8bf;
typedef __attribute__((ext_vector_type(8)))  float  v8f;

#define NTOK   4096
#define HD     64
#define HEADS  8
#define DIM    512
#define BATCH  4

__device__ __forceinline__ v8f wmma_bf16(v16bf a, v16bf b, v8f c) {
  return __builtin_amdgcn_wmma_f32_16x16x32_bf16(
      false, a, false, b, (short)0, c, false, false);
}

__device__ __forceinline__ v16bf cat8(v8bf lo, v8bf hi) {
  return __builtin_shufflevector(lo, hi, 0, 1, 2, 3, 4, 5, 6, 7,
                                 8, 9, 10, 11, 12, 13, 14, 15);
}

// A fragment (16x32 bf16) from row-major [M][K], row pitch ldm (16B-aligned)
__device__ __forceinline__ v16bf load_a_frag(const __bf16* __restrict__ base,
                                             int ldm) {
  const int lane = threadIdx.x & 31;
  const int g = lane >> 4, m = lane & 15;
  const __bf16* rp = base + (size_t)m * ldm + g * 8;
  v8bf lo = *(const v8bf*)rp;
  v8bf hi = *(const v8bf*)(rp + 16);
  return cat8(lo, hi);
}

// B fragment (32x16 bf16) from K-contiguous [N][K] layout, pitch ldn
__device__ __forceinline__ v16bf load_b_frag_t(const __bf16* __restrict__ base,
                                               int ldn) {
  const int lane = threadIdx.x & 31;
  const int g = lane >> 4, n = lane & 15;
  const __bf16* p = base + (size_t)n * ldn + g * 8;
  v8bf lo = *(const v8bf*)p;
  v8bf hi = *(const v8bf*)(p + 16);
  return cat8(lo, hi);
}

__device__ __forceinline__ float halfmax16(float v) {
#pragma unroll
  for (int m = 1; m < 16; m <<= 1) v = fmaxf(v, __shfl_xor(v, m, 32));
  return v;
}
__device__ __forceinline__ float halfsum16(float v) {
#pragma unroll
  for (int m = 1; m < 16; m <<= 1) v += __shfl_xor(v, m, 32);
  return v;
}

// ---- staging register blocks (one k-step of a 64x256 GEMM block) ----------
struct StageF32 {    // fp32 source (x / weights)
  float xr[32];      // thread t owns xs_t row t (32 K-values)
  float wr[8];       // thread owns wsh[t>>2][(t&3)*8 .. +7]
};

__device__ __forceinline__ void stage_load_f32(StageF32& R,
                                               const float* __restrict__ src,
                                               const float* __restrict__ w,
                                               int ob, int nb, int k0, int t) {
#pragma unroll
  for (int i = 0; i < 32; ++i)
    R.xr[i] = src[(size_t)(k0 + i) * NTOK + nb + t];
  const int wrow = t >> 2, wcol = (t & 3) * 8;
#pragma unroll
  for (int e = 0; e < 8; ++e)
    R.wr[e] = w[(size_t)(ob + wrow) * DIM + k0 + wcol + e];
}

__device__ __forceinline__ void stage_store(const StageF32& R,
                                            __bf16 (*xs)[40],
                                            __bf16 (*ws)[40], int t) {
#pragma unroll
  for (int ch = 0; ch < 4; ++ch) {
    v8bf pk;
#pragma unroll
    for (int e = 0; e < 8; ++e) pk[e] = (__bf16)R.xr[ch * 8 + e];
    *(v8bf*)&xs[t][ch * 8] = pk;
  }
  v8bf pkw;
#pragma unroll
  for (int e = 0; e < 8; ++e) pkw[e] = (__bf16)R.wr[e];
  *(v8bf*)&ws[t >> 2][(t & 3) * 8] = pkw;
}

// ===========================================================================
// Kernel 1: QKV GEMM + in-register RoPE.
// Wave (wm,wn) owns rows {wm*16+r+8g, +32} of the 64-row head block, so both
// halves of each rotary pair live in the same lane slot.
// ===========================================================================
__global__ __launch_bounds__(256)
void qkv_rope_kernel(const float* __restrict__ x,
                     const float* __restrict__ wqkv,
                     __bf16* __restrict__ Qw,
                     __bf16* __restrict__ Kw,
                     __bf16* __restrict__ Vt) {
  __shared__ __bf16 xs_t[2][256][40];  // B tile transposed [N][K], x2 buffers
  __shared__ __bf16 wsh[2][64][40];    // A tile [M][K]

  const int t = threadIdx.x;
  const int lane = t & 31, wave = t >> 5;
  const int wm = wave & 1, wn = wave >> 1;
  const int g = lane >> 4, cl = lane & 15;
  const int b = blockIdx.z;
  const int ob = blockIdx.y * 64;   // 64-aligned -> one head of one of q/k/v
  const int nb = blockIdx.x * 256;

  const float* xb = x + (size_t)b * DIM * NTOK;

  v8f zf = {0.f, 0.f, 0.f, 0.f, 0.f, 0.f, 0.f, 0.f};
  v8f acc0[4], acc1[4];
#pragma unroll
  for (int j = 0; j < 4; ++j) { acc0[j] = zf; acc1[j] = zf; }

  // prologue: stage k-step 0 into buffer 0
  {
    StageF32 R;
    stage_load_f32(R, xb, wqkv, ob, nb, 0, t);
    stage_store(R, xs_t[0], wsh[0], t);
  }
  __syncthreads();

  for (int ks = 0; ks < 16; ++ks) {
    const int p = ks & 1;
    const bool has_next = ks < 15;

    // issue next k-step's global loads first (latency hidden under WMMA)
    StageF32 R;
    if (has_next) stage_load_f32(R, xb, wqkv, ob, nb, (ks + 1) * 32, t);

    // compute from buffer p
    v16bf a0 = load_a_frag(&wsh[p][wm * 16][0], 40);
    v16bf a1 = load_a_frag(&wsh[p][wm * 16 + 32][0], 40);
    v16bf bf[4];
#pragma unroll
    for (int j = 0; j < 4; ++j)
      bf[j] = load_b_frag_t(&xs_t[p][wn * 64 + j * 16][0], 40);
#pragma unroll
    for (int j = 0; j < 4; ++j) {
      acc0[j] = wmma_bf16(a0, bf[j], acc0[j]);
      acc1[j] = wmma_bf16(a1, bf[j], acc1[j]);
    }

    // store prefetched data into the other buffer; single barrier per step
    if (has_next) stage_store(R, xs_t[p ^ 1], wsh[p ^ 1], t);
    __syncthreads();
  }

  // In-register RoPE epilogue. Lane slot (j, r): rows dd+r / dd+r+32,
  // column n = nb + wn*64 + j*16 + cl.
  const int sect = ob / DIM;            // 0=q, 1=k, 2=v
  const int h = (ob % DIM) / HD;
  const size_t bh = (size_t)b * HEADS + h;
  const float LOG1E4_OVER_Q = 0.5756462732485115f;  // ln(10000)/16
  const int dd = wm * 16 + 8 * g;

  if (sect == 2) {
    // V: [b][h][d][n]; 32B-contiguous runs along n
#pragma unroll
    for (int j = 0; j < 4; ++j)
#pragma unroll
      for (int r = 0; r < 8; ++r) {
        int n = nb + wn * 64 + j * 16 + cl;
        Vt[(bh * HD + dd + r) * NTOK + n]      = (__bf16)acc0[j][r];
        Vt[(bh * HD + dd + r + 32) * NTOK + n] = (__bf16)acc1[j][r];
      }
  } else {
    __bf16* dst = (sect == 0) ? Qw : Kw;
#pragma unroll
    for (int j = 0; j < 4; ++j)
#pragma unroll
      for (int r = 0; r < 8; ++r) {
        int d0 = dd + r;                 // rotary pair (d0, d0+32)
        int n = nb + wn * 64 + j * 16 + cl;
        float pos = (d0 < 16) ? (float)(n >> 6) : (float)(n & 63);
        int fi = d0 & 15;
        float ang = pos * __expf(-(float)fi * LOG1E4_OVER_Q);
        float sn, cn;
        __sincosf(ang, &sn, &cn);
        float x1 = acc0[j][r], x2 = acc1[j][r];
        __bf16* pq = dst + (bh * NTOK + n) * HD + d0;
        pq[0]  = (__bf16)(x1 * cn - x2 * sn);
        pq[32] = (__bf16)(x1 * sn + x2 * cn);
      }
  }
}

// ===========================================================================
// Kernel 2: flash attention; one wave = 16 query rows, online softmax.
// ===========================================================================
__global__ __launch_bounds__(256)
void attn_kernel(const __bf16* __restrict__ Qw,
                 const __bf16* __restrict__ Kw,
                 const __bf16* __restrict__ Vt,
                 __bf16* __restrict__ Ow) {
  __shared__ __bf16 Plds[8][16][40];  // per-wave P tile (16x32)

  const int t = threadIdx.x;
  const int lane = t & 31, wave = t >> 5;
  const int g = lane >> 4, cl = lane & 15;
  const int bh = blockIdx.y;  // b*8 + h
  const int b = bh >> 3, h = bh & 7;
  const int rowbase = blockIdx.x * 128 + wave * 16;

  const __bf16* Q = Qw + ((size_t)bh * NTOK + rowbase) * HD;
  const __bf16* K = Kw + (size_t)bh * NTOK * HD;
  const __bf16* V = Vt + (size_t)bh * HD * NTOK;  // [d][n]

  v16bf qa0 = load_a_frag(Q + 0, HD);
  v16bf qa1 = load_a_frag(Q + 32, HD);

  v8f zf = {0.f, 0.f, 0.f, 0.f, 0.f, 0.f, 0.f, 0.f};
  v8f o_acc[4];
#pragma unroll
  for (int d = 0; d < 4; ++d) o_acc[d] = zf;

  float mrow[8], lrow[8];
#pragma unroll
  for (int r = 0; r < 8; ++r) { mrow[r] = -3.0e38f; lrow[r] = 0.f; }

  const float scale = 0.125f;  // 64^-0.5

  for (int m0 = 0; m0 < NTOK; m0 += 32) {
    if (m0 + 32 < NTOK) {
      __builtin_prefetch(K + (size_t)(m0 + 32) * HD, 0, 1);
      __builtin_prefetch(V + m0 + 32, 0, 1);
    }
    // V fragments first: independent of softmax, hide global latency
    v16bf vb[4];
#pragma unroll
    for (int d = 0; d < 4; ++d)
      vb[d] = load_b_frag_t(V + (size_t)(d * 16) * NTOK + m0, NTOK);

    v16bf kb[4];
    kb[0] = load_b_frag_t(K + (size_t)m0 * HD + 0, HD);
    kb[1] = load_b_frag_t(K + (size_t)m0 * HD + 32, HD);
    kb[2] = load_b_frag_t(K + (size_t)(m0 + 16) * HD + 0, HD);
    kb[3] = load_b_frag_t(K + (size_t)(m0 + 16) * HD + 32, HD);
    v8f s0 = zf, s1 = zf;
    s0 = wmma_bf16(qa0, kb[0], s0);
    s0 = wmma_bf16(qa1, kb[1], s0);
    s1 = wmma_bf16(qa0, kb[2], s1);
    s1 = wmma_bf16(qa1, kb[3], s1);
    s0 = s0 * scale;
    s1 = s1 * scale;

    // online softmax over 32 columns
    float sc[8];
#pragma unroll
    for (int r = 0; r < 8; ++r) {
      float tmax = halfmax16(fmaxf(s0[r], s1[r]));
      float mnew = fmaxf(mrow[r], tmax);
      sc[r] = __expf(mrow[r] - mnew);
      mrow[r] = mnew;
      s0[r] = __expf(s0[r] - mnew);
      s1[r] = __expf(s1[r] - mnew);
      lrow[r] = lrow[r] * sc[r] + halfsum16(s0[r] + s1[r]);
    }
#pragma unroll
    for (int d = 0; d < 4; ++d)
#pragma unroll
      for (int r = 0; r < 8; ++r) o_acc[d][r] *= sc[r];

    // restage P (C layout -> row-major); wave-private, DS ops in-order
#pragma unroll
    for (int r = 0; r < 8; ++r) {
      Plds[wave][r + 8 * g][cl]      = (__bf16)s0[r];
      Plds[wave][r + 8 * g][16 + cl] = (__bf16)s1[r];
    }
    v16bf pa = load_a_frag(&Plds[wave][0][0], 40);

#pragma unroll
    for (int d = 0; d < 4; ++d)
      o_acc[d] = wmma_bf16(pa, vb[d], o_acc[d]);
  }

  // normalize; store O as [b][n][c] (c = h*64+d)
  float inv[8];
#pragma unroll
  for (int r = 0; r < 8; ++r) inv[r] = 1.0f / lrow[r];
#pragma unroll
  for (int d = 0; d < 4; ++d)
#pragma unroll
    for (int r = 0; r < 8; ++r) {
      int n = rowbase + r + 8 * g;
      Ow[((size_t)b * NTOK + n) * DIM + h * HD + d * 16 + cl] =
          (__bf16)(o_acc[d][r] * inv[r]);
    }
}

// ===========================================================================
// Kernel 3: output projection; same pipelined GEMM, bf16 source tile.
// ===========================================================================
__global__ __launch_bounds__(256)
void proj_kernel(const float* __restrict__ wproj,
                 const __bf16* __restrict__ Ow,
                 float* __restrict__ out) {
  __shared__ __bf16 xs_t[2][256][40];
  __shared__ __bf16 wsh[2][64][40];

  const int t = threadIdx.x;
  const int lane = t & 31, wave = t >> 5;
  const int wm = wave & 1, wn = wave >> 1;
  const int g = lane >> 4, cl = lane & 15;
  const int b = blockIdx.z;
  const int ob = blockIdx.y * 64;
  const int nb = blockIdx.x * 256;

  const __bf16* Ob = Ow + (size_t)b * NTOK * DIM;  // [n][c]

  v8f zf = {0.f, 0.f, 0.f, 0.f, 0.f, 0.f, 0.f, 0.f};
  v8f acc0[4], acc1[4];
#pragma unroll
  for (int j = 0; j < 4; ++j) { acc0[j] = zf; acc1[j] = zf; }

  // thread t owns xs_t row t: 4x 16B vector loads (already bf16)
  auto load_x = [&](v8bf* xr, int k0) {
#pragma unroll
    for (int ch = 0; ch < 4; ++ch)
      xr[ch] = *(const v8bf*)&Ob[(size_t)(nb + t) * DIM + k0 + ch * 8];
  };
  auto load_w = [&](float* wr, int k0) {
    const int wrow = t >> 2, wcol = (t & 3) * 8;
#pragma unroll
    for (int e = 0; e < 8; ++e)
      wr[e] = wproj[(size_t)(ob + wrow) * DIM + k0 + wcol + e];
  };
  auto store_tiles = [&](const v8bf* xr, const float* wr, int p) {
#pragma unroll
    for (int ch = 0; ch < 4; ++ch)
      *(v8bf*)&xs_t[p][t][ch * 8] = xr[ch];
    v8bf pkw;
#pragma unroll
    for (int e = 0; e < 8; ++e) pkw[e] = (__bf16)wr[e];
    *(v8bf*)&wsh[p][t >> 2][(t & 3) * 8] = pkw;
  };

  {
    v8bf xr[4]; float wr[8];
    load_x(xr, 0);
    load_w(wr, 0);
    store_tiles(xr, wr, 0);
  }
  __syncthreads();

  for (int ks = 0; ks < 16; ++ks) {
    const int p = ks & 1;
    const bool has_next = ks < 15;

    v8bf xr[4]; float wr[8];
    if (has_next) { load_x(xr, (ks + 1) * 32); load_w(wr, (ks + 1) * 32); }

    v16bf a0 = load_a_frag(&wsh[p][wm * 16][0], 40);
    v16bf a1 = load_a_frag(&wsh[p][wm * 16 + 32][0], 40);
    v16bf bf[4];
#pragma unroll
    for (int j = 0; j < 4; ++j)
      bf[j] = load_b_frag_t(&xs_t[p][wn * 64 + j * 16][0], 40);
#pragma unroll
    for (int j = 0; j < 4; ++j) {
      acc0[j] = wmma_bf16(a0, bf[j], acc0[j]);
      acc1[j] = wmma_bf16(a1, bf[j], acc1[j]);
    }

    if (has_next) store_tiles(xr, wr, p ^ 1);
    __syncthreads();
  }

  // direct fp32 stores
#pragma unroll
  for (int j = 0; j < 4; ++j)
#pragma unroll
    for (int r = 0; r < 8; ++r) {
      int row0 = ob + wm * 16 + r + 8 * g;
      int col = nb + wn * 64 + j * 16 + cl;
      float* o = out + (size_t)b * DIM * NTOK + col;
      o[(size_t)row0 * NTOK]        = acc0[j][r];
      o[(size_t)(row0 + 32) * NTOK] = acc1[j][r];
    }
}

// ===========================================================================
extern "C" void kernel_launch(void* const* d_in, const int* in_sizes, int n_in,
                              void* d_out, int out_size, void* d_ws,
                              size_t ws_size, hipStream_t stream) {
  (void)in_sizes; (void)n_in; (void)out_size; (void)ws_size;
  const float* x     = (const float*)d_in[0];
  const float* wqkv  = (const float*)d_in[1];
  const float* wproj = (const float*)d_in[2];
  float* out = (float*)d_out;

  const size_t per = (size_t)BATCH * HEADS * NTOK * HD;  // 8.39M bf16 elems
  __bf16* Qw = (__bf16*)d_ws;
  __bf16* Kw = Qw + per;
  __bf16* Vt = Kw + per;   // [b][h][d][n]
  __bf16* Ow = Vt + per;   // [b][n][c]

  dim3 blk(256);
  qkv_rope_kernel<<<dim3(16, 24, 4), blk, 0, stream>>>(x, wqkv, Qw, Kw, Vt);
  attn_kernel<<<dim3(32, 32, 1), blk, 0, stream>>>(Qw, Kw, Vt, Ow);
  proj_kernel<<<dim3(16, 8, 4), blk, 0, stream>>>(wproj, Ow, out);
}
